// BabiUTransformer_74406013436555
// MI455X (gfx1250) — matrix operand order repaired and checked by reading, not compiled, not run on hardware
//
#include <hip/hip_runtime.h>
#include <math.h>

typedef __attribute__((ext_vector_type(16))) _Float16 v16h;
typedef __attribute__((ext_vector_type(8)))  float    v8f;

#define ML   18176      // B*71 rows
#define LSEQ 71
#define HDIM 512
#define FSD  2048
#define VOC  32000
#define NBATCH 256

// ---------------------------------------------------------------------------
// CDNA5 async global->LDS copy (ASYNCcnt-tracked), 16 bytes per lane.
// Flat pointers to LDS carry the LDS byte offset in the low 32 bits
// (aperture lives in addr[63:32], ISA 10.2), so truncation gives VDST.
// ---------------------------------------------------------------------------
__device__ __forceinline__ void async_copy_b128(const void* gptr, void* lptr) {
    unsigned lds = (unsigned)(size_t)lptr;
    unsigned long long ga = (unsigned long long)(size_t)gptr;
    asm volatile("global_load_async_to_lds_b128 %0, %1, off"
                 :: "v"(lds), "v"(ga) : "memory");
}
__device__ __forceinline__ void wait_async0() {
    asm volatile("s_wait_asynccnt 0x0" ::: "memory");
}

// ---------------------------------------------------------------------------
// Generic tiled WMMA GEMM:  C[M,N] (+bias[n]) (+res[row,n]) (relu?) = A @ B
// A:  f16 [M,Kd] row-major (optionally row-shifted per conv tap, zero pad)
// Bw: f16 [ntaps][N,Kd]  (N-major, pre-transposed at pack time)
// Writes f32 (Cf) and/or f16 (Ch).
// Block tile 128x128, 256 threads = 8 waves (4M x 2N), wave tile 32x64.
// K staged 64 at a time (two 32-K WMMA sub-steps per stage).
// ---------------------------------------------------------------------------
__global__ __launch_bounds__(256)
void gemm_f16(const _Float16* __restrict__ A, const _Float16* __restrict__ Bw,
              float* __restrict__ Cf, _Float16* __restrict__ Ch,
              const float* __restrict__ bias, const float* __restrict__ res,
              int M, int N, int Kd, int L, int ntaps, int relu)
{
    __shared__ _Float16 As[128 * 72];   // [m][k], row stride 72 halves (144B, 16B-aligned)
    __shared__ _Float16 Bs[128 * 72];   // [n][k]

    const int tid    = threadIdx.x;
    const int lane   = tid & 31;
    const int w      = tid >> 5;
    const int wm     = w & 3;           // 0..3 -> M offset wm*32
    const int wn     = w >> 2;          // 0..1 -> N offset wn*64
    const int laneLo = lane & 15;
    const int hi     = lane >> 4;       // 0/1
    const int mt     = blockIdx.y;
    const int nb     = blockIdx.x;

    v8f acc[2][4] = {};
    const int nk = Kd >> 6;

    // per-thread staging geometry: 4 chunks of 8 halves per 128x64 tile
    int rowIdx[4], segOf[4];
    #pragma unroll
    for (int i = 0; i < 4; ++i) {
        int c = tid + i * 256;
        rowIdx[i] = c >> 3;             // 0..127
        segOf[i]  = (c & 7) << 3;       // 0..56
    }

    for (int tap = 0; tap < ntaps; ++tap) {
        const int d = (ntaps == 3) ? (tap - 1) : 0;
        const _Float16* Bt = Bw + (size_t)tap * N * Kd;

        // hoist shift validity + base pointers out of the k loop
        const _Float16* aSrc[4];
        const _Float16* bSrc[4];
        bool aVal[4];
        #pragma unroll
        for (int i = 0; i < 4; ++i) {
            int gr   = mt * 128 + rowIdx[i];
            int l    = gr % L;
            aVal[i]  = (unsigned)(l + d) < (unsigned)L;
            aSrc[i]  = A  + (size_t)(gr + d) * Kd + segOf[i];
            bSrc[i]  = Bt + (size_t)(nb * 128 + rowIdx[i]) * Kd + segOf[i];
        }

        for (int kt = 0; kt < nk; ++kt) {
            const int k0 = kt << 6;
            // ---- async-stage B tile (weights, always valid) ----
            #pragma unroll
            for (int i = 0; i < 4; ++i)
                async_copy_b128(bSrc[i] + k0, &Bs[rowIdx[i] * 72 + segOf[i]]);
            // ---- stage A tile, zero-fill shifted-out rows ----
            #pragma unroll
            for (int i = 0; i < 4; ++i) {
                uint4 val = make_uint4(0u, 0u, 0u, 0u);
                if (aVal[i]) {
                    val = *(const uint4*)(aSrc[i] + k0);
                    if (kt + 1 < nk) __builtin_prefetch(aSrc[i] + k0 + 64, 0, 1);
                }
                *(uint4*)(&As[rowIdx[i] * 72 + segOf[i]]) = val;
            }
            wait_async0();
            __syncthreads();

            // ---- two 32-K WMMA sub-steps (ISA 7.12.2 lane layouts) ----
            union Frag { v16h v; unsigned u[8]; };
            #pragma unroll
            for (int ks = 0; ks < 2; ++ks) {
                Frag af[2], bf[4];
                #pragma unroll
                for (int fm = 0; fm < 2; ++fm) {
                    int m = wm * 32 + fm * 16 + laneLo;
                    const unsigned* row = (const unsigned*)&As[m * 72];
                    #pragma unroll
                    for (int g = 0; g < 8; ++g)
                        af[fm].u[g] = row[ks * 16 + (g < 4 ? g : g + 4) + 4 * hi];
                }
                #pragma unroll
                for (int fn = 0; fn < 4; ++fn) {
                    int n = wn * 64 + fn * 16 + laneLo;
                    const unsigned* row = (const unsigned*)&Bs[n * 72];
                    #pragma unroll
                    for (int g = 0; g < 8; ++g)
                        bf[fn].u[g] = row[ks * 16 + (g < 4 ? g : g + 4) + 4 * hi];
                }
                #pragma unroll
                for (int fm = 0; fm < 2; ++fm)
                    #pragma unroll
                    for (int fn = 0; fn < 4; ++fn)
                        acc[fm][fn] = __builtin_amdgcn_wmma_f32_16x16x32_f16(
                            false, af[fm].v, false, bf[fn].v,
                            (short)0, acc[fm][fn], false, false);
            }
            __syncthreads();
        }
    }

    // ---- epilogue ----
    #pragma unroll
    for (int fm = 0; fm < 2; ++fm) {
        #pragma unroll
        for (int fn = 0; fn < 4; ++fn) {
            int col = nb * 128 + wn * 64 + fn * 16 + laneLo;
            float bv = bias ? bias[col] : 0.f;
            #pragma unroll
            for (int e = 0; e < 8; ++e) {
                int row = mt * 128 + wm * 32 + fm * 16 + e + hi * 8;
                float v = acc[fm][fn][e] + bv;
                size_t oi = (size_t)row * N + col;
                if (res)  v += res[oi];
                if (relu) v = v > 0.f ? v : 0.f;
                if (Cf) Cf[oi] = v;
                if (Ch) Ch[oi] = (_Float16)v;
            }
        }
    }
}

// ---------------------------------------------------------------------------
// small helpers
// ---------------------------------------------------------------------------
__device__ __forceinline__ float blockReduceSum256(float v, float* red) {
    int t = threadIdx.x;
    red[t] = v; __syncthreads();
    for (int s = 128; s > 0; s >>= 1) {
        if (t < s) red[t] += red[t + s];
        __syncthreads();
    }
    float r = red[0]; __syncthreads();
    return r;
}

__device__ __forceinline__ float tsig(int pos, int c) {
    const float log_inc = 9.210340371976184f / 255.f;
    int i = c & 255;
    float inv = expf(-(float)i * log_inc);
    float sc  = (float)pos * inv;
    return (c < 256) ? sinf(sc) : cosf(sc);
}

// ---------------------------------------------------------------------------
// weight conversion / packing (all B operands packed N-major: out[n][k])
// ---------------------------------------------------------------------------
__global__ void f32_to_f16_k(const float* __restrict__ in, _Float16* __restrict__ out, int n) {
    int i = blockIdx.x * 256 + threadIdx.x;
    if (i < n) out[i] = (_Float16)in[i];
}
// dense weight [K,N] -> out[n*K+k]
__global__ void pack_T_k(const float* __restrict__ in, _Float16* __restrict__ out,
                         int K, int N) {
    int i = blockIdx.x * 256 + threadIdx.x;
    if (i >= K * N) return;
    int n = i / K, k = i % K;
    out[i] = (_Float16)in[(size_t)k * N + n];
}
// torch conv weight [Cout,Cin,3] -> out[tap][co][ci]
__global__ void pack_convT_k(const float* __restrict__ w, _Float16* __restrict__ out,
                             int Cout, int Cin) {
    int i = blockIdx.x * 256 + threadIdx.x;
    int tot = 3 * Cout * Cin;
    if (i >= tot) return;
    int tap = i / (Cout * Cin);
    int r   = i % (Cout * Cin);
    int co  = r / Cin;
    int ci  = r % Cin;
    out[i] = (_Float16)w[((size_t)co * Cin + ci) * 3 + tap];
}

// ---------------------------------------------------------------------------
// embedding sum (story rows 0..69, query row 70) -> f16 A matrix [ML, H]
// ---------------------------------------------------------------------------
__global__ void embed_k(const int* __restrict__ story, const int* __restrict__ query,
                        const float* __restrict__ emb, const float* __restrict__ maskp,
                        _Float16* __restrict__ out) {
    int i = blockIdx.x * 256 + threadIdx.x;
    if (i >= ML * HDIM) return;
    int c = i & 511, r = i >> 9;
    int b = r / LSEQ, m = r % LSEQ;
    float acc = 0.f;
    if (m < 70) {
        const int* tk = story + (size_t)(b * 70 + m) * 11;
        #pragma unroll
        for (int s = 0; s < 11; ++s)
            acc += emb[(size_t)tk[s] * HDIM + c] * maskp[s * HDIM + c];
    } else {
        const int* tk = query + (size_t)b * 11;
        #pragma unroll
        for (int s = 0; s < 11; ++s)
            acc += emb[(size_t)tk[s] * HDIM + c] * maskp[s * HDIM + c];
    }
    out[i] = (_Float16)acc;
}

// s = state + TIME_SIG[l] + POS_SIG[t]
__global__ void compute_s_k(const float* __restrict__ state, float* __restrict__ s, int t) {
    int i = blockIdx.x * 256 + threadIdx.x;
    if (i >= ML * HDIM) return;
    int c = i & 511, row = i >> 9, l = row % LSEQ;
    s[i] = state[i] + tsig(l, c) + tsig(t, c);
}

// ACT halting, one block per (b,l) row
__global__ __launch_bounds__(256)
void halt_row_k(const float* __restrict__ s, const float* __restrict__ pw,
                const float* __restrict__ pb,
                const float* __restrict__ hp, const float* __restrict__ rem,
                const float* __restrict__ nu,
                float* __restrict__ hpN, float* __restrict__ remN,
                float* __restrict__ nuN, float* __restrict__ uw) {
    __shared__ float red[256];
    int row = blockIdx.x, t = threadIdx.x;
    const float* sr = s + (size_t)row * HDIM;
    float part = sr[t] * pw[t] + sr[t + 256] * pw[t + 256];
    float dot = blockReduceSum256(part, red);
    if (t == 0) {
        float pr   = 1.f / (1.f + expf(-(dot + pb[0])));
        float hpv  = hp[row], remv = rem[row], nuv = nu[row];
        float still = (hpv < 1.0f) ? 1.f : 0.f;
        float tot  = hpv + pr * still;
        float nh   = ((tot > 0.9f) ? 1.f : 0.f) * still;
        float st2  = ((tot <= 0.9f) ? 1.f : 0.f) * still;
        float hp2  = hpv + pr * st2;
        float rem2 = remv + nh * (1.f - hp2);
        hp2 += nh * rem2;
        float nu2  = nuv + st2 + nh;
        hpN[row]  = hp2;
        remN[row] = rem2;
        nuN[row]  = nu2;
        uw[row]   = pr * st2 + nh * rem2;
    }
}

// LayerNorm (torch ddof=1, sd + EPS), f32 in, f16 out
__global__ __launch_bounds__(256)
void layernorm_k(const float* __restrict__ x, const float* __restrict__ g,
                 const float* __restrict__ b, _Float16* __restrict__ out) {
    __shared__ float red[256];
    int row = blockIdx.x, t = threadIdx.x;
    const float* xr = x + (size_t)row * HDIM;
    float v0 = xr[t], v1 = xr[t + 256];
    float mu = blockReduceSum256(v0 + v1, red) * (1.f / 512.f);
    float d0 = v0 - mu, d1 = v1 - mu;
    float ss = blockReduceSum256(d0 * d0 + d1 * d1, red);
    float sd = sqrtf(ss / 511.f);
    float inv = 1.f / (sd + 1e-6f);
    out[(size_t)row * HDIM + t]       = (_Float16)(g[t] * d0 * inv + b[t]);
    out[(size_t)row * HDIM + t + 256] = (_Float16)(g[t + 256] * d1 * inv + b[t + 256]);
}

// attention: one block per (batch, head); q pre-scaled by dk^-0.5
__global__ __launch_bounds__(96)
void attn_k(const float* __restrict__ q, const float* __restrict__ k,
            const float* __restrict__ v, _Float16* __restrict__ ctx) {
    __shared__ float qs[LSEQ * 64], ks[LSEQ * 64], vs[LSEQ * 64];
    int b = blockIdx.x >> 3, h = blockIdx.x & 7;
    size_t base = ((size_t)b * LSEQ) * HDIM + h * 64;
    for (int i = threadIdx.x; i < LSEQ * 64; i += 96) {
        int l = i >> 6, dd = i & 63;
        size_t gi = base + (size_t)l * HDIM + dd;
        qs[i] = q[gi] * 0.125f;
        ks[i] = k[gi];
        vs[i] = v[gi];
    }
    __syncthreads();
    int i = threadIdx.x;
    if (i < LSEQ) {
        const float* qi = qs + i * 64;
        float mx = -1e30f;
        for (int j = 0; j < LSEQ; ++j) {
            const float* kj = ks + j * 64;
            float d2 = 0.f;
            #pragma unroll 8
            for (int dd = 0; dd < 64; ++dd) d2 += qi[dd] * kj[dd];
            mx = fmaxf(mx, d2);
        }
        float acc[64];
        #pragma unroll
        for (int dd = 0; dd < 64; ++dd) acc[dd] = 0.f;
        float sum = 0.f;
        for (int j = 0; j < LSEQ; ++j) {
            const float* kj = ks + j * 64;
            float d2 = 0.f;
            #pragma unroll 8
            for (int dd = 0; dd < 64; ++dd) d2 += qi[dd] * kj[dd];
            float e = expf(d2 - mx);
            sum += e;
            const float* vj = vs + j * 64;
            #pragma unroll 8
            for (int dd = 0; dd < 64; ++dd) acc[dd] += e * vj[dd];
        }
        float inv = 1.f / sum;
        for (int dd = 0; dd < 64; ++dd)
            ctx[base + (size_t)i * HDIM + dd] = (_Float16)(acc[dd] * inv);
    }
}

// ACT gating flag
__global__ void flag_zero_k(int* f) { *f = 0; }
__global__ void flag_any_k(const float* __restrict__ hp, const float* __restrict__ nu, int* f) {
    int i = blockIdx.x * 256 + threadIdx.x;
    if (i >= ML) return;
    if (hp[i] < 0.9f && nu[i] < 6.0f) atomicOr(f, 1);
}

// carry = where(active, new, old)
__global__ void commit_k(const int* __restrict__ flag, const float* __restrict__ s2,
                         float* __restrict__ state, float* __restrict__ prev,
                         const float* __restrict__ uw,
                         float* __restrict__ hp, const float* __restrict__ hpN,
                         float* __restrict__ rem, const float* __restrict__ remN,
                         float* __restrict__ nu, const float* __restrict__ nuN) {
    if (!*flag) return;
    int i = blockIdx.x * 256 + threadIdx.x;
    if (i >= ML * HDIM) return;
    int row = i >> 9, c = i & 511;
    float v = s2[i];
    state[i] = v;
    float u = uw[row];
    prev[i] = v * u + prev[i] * (1.f - u);
    if (c == 0) { hp[row] = hpN[row]; rem[row] = remN[row]; nu[row] = nuN[row]; }
}

// pooled[b,c] = mean over l of prev  (f16 out for final GEMM)
__global__ void pool_k(const float* __restrict__ prev, _Float16* __restrict__ out) {
    int i = blockIdx.x * 256 + threadIdx.x;
    if (i >= NBATCH * HDIM) return;
    int b = i >> 9, c = i & 511;
    float acc = 0.f;
    for (int l = 0; l < LSEQ; ++l)
        acc += prev[((size_t)(b * LSEQ + l)) * HDIM + c];
    out[i] = (_Float16)(acc * (1.f / (float)LSEQ));
}

__global__ __launch_bounds__(256)
void softmax_k(const float* __restrict__ in, float* __restrict__ out, int N) {
    __shared__ float red[256];
    int row = blockIdx.x, t = threadIdx.x;
    const float* x = in + (size_t)row * N;
    float* y = out + (size_t)row * N;
    float mx = -1e30f;
    for (int c = t; c < N; c += 256) mx = fmaxf(mx, x[c]);
    red[t] = mx; __syncthreads();
    for (int s = 128; s > 0; s >>= 1) {
        if (t < s) red[t] = fmaxf(red[t], red[t + s]);
        __syncthreads();
    }
    mx = red[0]; __syncthreads();
    float sum = 0.f;
    for (int c = t; c < N; c += 256) sum += expf(x[c] - mx);
    sum = blockReduceSum256(sum, red);
    float inv = 1.f / sum;
    for (int c = t; c < N; c += 256) y[c] = expf(x[c] - mx) * inv;
}

__global__ void tail_k(const float* __restrict__ rem, const float* __restrict__ nu,
                       float* __restrict__ out) {
    int i = blockIdx.x * 256 + threadIdx.x;
    if (i >= ML) return;
    out[i]      = rem[i];
    out[ML + i] = nu[i];
}

// ---------------------------------------------------------------------------
extern "C" void kernel_launch(void* const* d_in, const int* in_sizes, int n_in,
                              void* d_out, int out_size, void* d_ws, size_t ws_size,
                              hipStream_t stream) {
    (void)in_sizes; (void)n_in; (void)out_size; (void)ws_size;
    const int*   story = (const int*)d_in[0];
    const int*   query = (const int*)d_in[1];
    const float* emb   = (const float*)d_in[2];
    const float* maskp = (const float*)d_in[3];
    const float* projw = (const float*)d_in[4];
    const float* ln1g  = (const float*)d_in[5];
    const float* ln1b  = (const float*)d_in[6];
    const float* wq    = (const float*)d_in[7];
    const float* wk    = (const float*)d_in[8];
    const float* wv    = (const float*)d_in[9];
    const float* wo    = (const float*)d_in[10];
    const float* c1w   = (const float*)d_in[11];
    const float* c1b   = (const float*)d_in[12];
    const float* c2w   = (const float*)d_in[13];
    const float* c2b   = (const float*)d_in[14];
    const float* ln2g  = (const float*)d_in[15];
    const float* ln2b  = (const float*)d_in[16];
    const float* pw    = (const float*)d_in[17];
    const float* pb    = (const float*)d_in[18];
    const float* outb  = (const float*)d_in[19];
    float* outp = (float*)d_out;

    // ---- workspace carve (bump allocator, 256B aligned) ----
    char* p = (char*)d_ws;
    auto alloc = [&](size_t bytes) -> void* {
        void* r = (void*)p;
        p += (bytes + 255) & ~(size_t)255;
        return r;
    };
    const size_t MH  = (size_t)ML * HDIM;
    _Float16* proj_h = (_Float16*)alloc(512 * 512 * 2);
    _Float16* wq_h   = (_Float16*)alloc(512 * 512 * 2);
    _Float16* wk_h   = (_Float16*)alloc(512 * 512 * 2);
    _Float16* wv_h   = (_Float16*)alloc(512 * 512 * 2);
    _Float16* wo_h   = (_Float16*)alloc(512 * 512 * 2);
    _Float16* c1w_h  = (_Float16*)alloc((size_t)3 * 512 * 2048 * 2);
    _Float16* c2w_h  = (_Float16*)alloc((size_t)3 * 2048 * 512 * 2);
    _Float16* embT_h = (_Float16*)alloc((size_t)HDIM * VOC * 2);   // emb as [V][E] = [N][K]
    float* state = (float*)alloc(MH * 4);
    float* sbuf  = (float*)alloc(MH * 4);          // s, later reused as s2
    float* prev  = (float*)alloc(MH * 4);
    float* x1    = (float*)alloc(MH * 4);
    float* qb    = (float*)alloc(MH * 4);
    float* kb    = (float*)alloc(MH * 4);
    float* vb    = (float*)alloc(MH * 4);
    _Float16* xn_h  = (_Float16*)alloc(MH * 2);    // also embedding A matrix
    _Float16* ctx_h = (_Float16*)alloc(MH * 2);
    _Float16* y1_h  = (_Float16*)alloc((size_t)ML * FSD * 2);
    float* hp   = (float*)alloc(ML * 4);
    float* rem  = (float*)alloc(ML * 4);
    float* nu   = (float*)alloc(ML * 4);
    float* hpN  = (float*)alloc(ML * 4);
    float* remN = (float*)alloc(ML * 4);
    float* nuN  = (float*)alloc(ML * 4);
    float* uwB  = (float*)alloc(ML * 4);
    _Float16* pooled_h = (_Float16*)alloc(NBATCH * HDIM * 2);
    int* flag = (int*)alloc(256);

    const int EW = 256;
    #define GB(n) dim3(((n) + EW - 1) / EW)

    // ---- pack weights (all B operands N-major) ----
    pack_T_k<<<GB(262144), EW, 0, stream>>>(projw, proj_h, 512, 512);
    pack_T_k<<<GB(262144), EW, 0, stream>>>(wq, wq_h, 512, 512);
    pack_T_k<<<GB(262144), EW, 0, stream>>>(wk, wk_h, 512, 512);
    pack_T_k<<<GB(262144), EW, 0, stream>>>(wv, wv_h, 512, 512);
    pack_T_k<<<GB(262144), EW, 0, stream>>>(wo, wo_h, 512, 512);
    pack_convT_k<<<GB(3 * 2048 * 512), EW, 0, stream>>>(c1w, c1w_h, 2048, 512);
    pack_convT_k<<<GB(3 * 512 * 2048), EW, 0, stream>>>(c2w, c2w_h, 512, 2048);
    f32_to_f16_k<<<GB(VOC * HDIM), EW, 0, stream>>>(emb, embT_h, VOC * HDIM);

    // ---- embed + input projection -> initial state ----
    embed_k<<<GB(MH), EW, 0, stream>>>(story, query, emb, maskp, xn_h);
    gemm_f16<<<dim3(HDIM / 128, ML / 128), 256, 0, stream>>>(
        xn_h, proj_h, state, nullptr, nullptr, nullptr, ML, HDIM, HDIM, LSEQ, 1, 0);

    hipMemsetAsync(prev, 0, MH * 4, stream);
    hipMemsetAsync(hp,   0, ML * 4, stream);
    hipMemsetAsync(rem,  0, ML * 4, stream);
    hipMemsetAsync(nu,   0, ML * 4, stream);

    // ---- ACT loop (NL = 6), gated by device-side flag ----
    for (int t = 0; t < 6; ++t) {
        flag_zero_k<<<1, 1, 0, stream>>>(flag);
        flag_any_k<<<GB(ML), EW, 0, stream>>>(hp, nu, flag);
        compute_s_k<<<GB(MH), EW, 0, stream>>>(state, sbuf, t);
        halt_row_k<<<ML, 256, 0, stream>>>(sbuf, pw, pb, hp, rem, nu, hpN, remN, nuN, uwB);
        // encoder layer
        layernorm_k<<<ML, 256, 0, stream>>>(sbuf, ln1g, ln1b, xn_h);
        gemm_f16<<<dim3(4, ML / 128), 256, 0, stream>>>(
            xn_h, wq_h, qb, nullptr, nullptr, nullptr, ML, HDIM, HDIM, LSEQ, 1, 0);
        gemm_f16<<<dim3(4, ML / 128), 256, 0, stream>>>(
            xn_h, wk_h, kb, nullptr, nullptr, nullptr, ML, HDIM, HDIM, LSEQ, 1, 0);
        gemm_f16<<<dim3(4, ML / 128), 256, 0, stream>>>(
            xn_h, wv_h, vb, nullptr, nullptr, nullptr, ML, HDIM, HDIM, LSEQ, 1, 0);
        attn_k<<<NBATCH * 8, 96, 0, stream>>>(qb, kb, vb, ctx_h);
        gemm_f16<<<dim3(4, ML / 128), 256, 0, stream>>>(
            ctx_h, wo_h, x1, nullptr, nullptr, sbuf, ML, HDIM, HDIM, LSEQ, 1, 0);
        layernorm_k<<<ML, 256, 0, stream>>>(x1, ln2g, ln2b, xn_h);
        gemm_f16<<<dim3(FSD / 128, ML / 128), 256, 0, stream>>>(          // conv1 + relu
            xn_h, c1w_h, nullptr, y1_h, c1b, nullptr, ML, FSD, HDIM, LSEQ, 3, 1);
        gemm_f16<<<dim3(HDIM / 128, ML / 128), 256, 0, stream>>>(         // conv2 + residual
            y1_h, c2w_h, sbuf, nullptr, c2b, x1, ML, HDIM, FSD, LSEQ, 3, 0);
        commit_k<<<GB(MH), EW, 0, stream>>>(flag, sbuf, state, prev, uwB,
                                            hp, hpN, rem, remN, nu, nuN);
    }

    // ---- pool, vocab GEMM, softmax, tails ----
    pool_k<<<GB(NBATCH * HDIM), EW, 0, stream>>>(prev, pooled_h);
    gemm_f16<<<dim3(VOC / 128, NBATCH / 128), 256, 0, stream>>>(
        pooled_h, embT_h, outp, nullptr, outb, nullptr, NBATCH, VOC, HDIM, LSEQ, 1, 0);
    softmax_k<<<NBATCH, 256, 0, stream>>>(outp, outp + (size_t)NBATCH * VOC, VOC);
    tail_k<<<GB(ML), EW, 0, stream>>>(rem, nu, outp + (size_t)2 * NBATCH * VOC);
}
